// StochasticDurationPredictor_58514634441269
// MI455X (gfx1250) — matrix operand order repaired and checked by reading, not compiled, not run on hardware
//
#include <hip/hip_runtime.h>
#include <hip/hip_bf16.h>
#include <cstddef>

// ---------------------------------------------------------------------------
// StochasticDurationPredictor inverse pass for MI455X (gfx1250, wave32).
// Heavy ops: 14x [192x192 @ 65536-col] GEMMs -> bf16 WMMA (v_wmma_f32_16x16x32_bf16)
// with fused bias / channel-LayerNorm / GELU / residual epilogues.
// ---------------------------------------------------------------------------

typedef __attribute__((ext_vector_type(8)))  float   v8f;
typedef __attribute__((ext_vector_type(8)))  __bf16  bf16x8;
typedef __attribute__((ext_vector_type(16))) __bf16  bf16x16;

#define BB   64
#define CC   192
#define TT   1024
#define NCOL (BB * TT)          // 65536
#define KPAD 208                // bf16 elems; 208*2 = 416 B, multiple of 32B

__device__ __forceinline__ float geluf(float x) {
    return 0.5f * x * (1.0f + erff(x * 0.70710678118654752440f));
}
__device__ __forceinline__ float softplusf(float x) {
    return (x > 20.0f) ? x : log1pf(expf(x));
}

// ---------------------------------------------------------------------------
// GEMM: out[b,m,t] = epilogue( sum_k W[m,k] * X[b,k,t] )
// Block: 256 threads = 8 waves; tile M = MT*16 (<=192), N = 128 cols, K = 192.
// MODE 0: + bias[m] + cond[b,m]                       (pre conv)
// MODE 1: v=acc+bias; LN over channels; GELU; +resid  (dds pointwise conv)
// MODE 2: (acc+bias) * outmask                        (proj / cf_proj)
// ---------------------------------------------------------------------------
template <int MT, int MODE>
__global__ __launch_bounds__(256) void gemm192(
    const float* __restrict__ W, int Mrows,
    const float* __restrict__ X,
    const float* __restrict__ bias,
    const float* __restrict__ cond,
    const float* __restrict__ resid,
    const float* __restrict__ lng, const float* __restrict__ lnb,
    const float* __restrict__ inmask, const float* __restrict__ outmask,
    float* __restrict__ out)
{
    extern __shared__ char smem[];
    __bf16* Ws = (__bf16*)smem;                                   // [MT*16][KPAD]
    __bf16* Xs = (__bf16*)(smem + (size_t)MT * 16 * KPAD * 2);    // [128][KPAD]

    const int tid  = threadIdx.x;
    const int col0 = blockIdx.x * 128;
    const int b    = col0 / TT;
    const int t0   = col0 % TT;

    // Fill weights (f32 -> bf16), zero-pad rows >= Mrows.
    for (int idx = tid; idx < MT * 16 * CC; idx += 256) {
        int o = idx / CC, k = idx - o * CC;
        float wv = (o < Mrows) ? W[o * CC + k] : 0.0f;
        Ws[o * KPAD + k] = (__bf16)wv;
    }
    // Fill X tile transposed to [n][k] (global read coalesced along t).
    for (int idx = tid; idx < CC * 128; idx += 256) {
        int k = idx >> 7, n = idx & 127;
        float xv = X[((size_t)b * CC + k) * TT + t0 + n];
        if (inmask) xv *= inmask[b * TT + t0 + n];
        Xs[n * KPAD + k] = (__bf16)xv;
    }
    __syncthreads();

    const int lane = tid & 31;
    const int wv_  = tid >> 5;        // wave id 0..7 -> 16-col strip
    const int cl   = lane & 15;
    const int hi   = lane >> 4;       // 0/1
    const int ncol = wv_ * 16 + cl;   // column within tile

    v8f acc[MT];
#pragma unroll
    for (int mt = 0; mt < MT; ++mt)
#pragma unroll
        for (int r = 0; r < 8; ++r) acc[mt][r] = 0.0f;

#pragma unroll
    for (int kt = 0; kt < 6; ++kt) {
        const int kk = kt * 32;
        // B fragment: lane holds 16 consecutive K at column ncol, K-block = hi*16
        bf16x8 bl = *(const bf16x8*)&Xs[ncol * KPAD + kk + hi * 16];
        bf16x8 bh = *(const bf16x8*)&Xs[ncol * KPAD + kk + hi * 16 + 8];
        bf16x16 bfrag = __builtin_shufflevector(bl, bh,
            0,1,2,3,4,5,6,7,8,9,10,11,12,13,14,15);
#pragma unroll
        for (int mt = 0; mt < MT; ++mt) {
            const int o  = mt * 16 + cl;
            const int kb = kk + hi * 16;
            // A fragment: VGPR0-3 = K kb..kb+7, VGPR4-7 = K kb+8..kb+15
            bf16x8 a0 = *(const bf16x8*)&Ws[o * KPAD + kb];
            bf16x8 a1 = *(const bf16x8*)&Ws[o * KPAD + kb + 8];
            bf16x16 afrag = __builtin_shufflevector(a0, a1,
                0,1,2,3,4,5,6,7,8,9,10,11,12,13,14,15);
            acc[mt] = __builtin_amdgcn_wmma_f32_16x16x32_bf16(
                false, afrag, false, bfrag, (short)0, acc[mt], false, false);
        }
    }

    const int t = t0 + ncol;

    if constexpr (MODE == 0) {
#pragma unroll
        for (int mt = 0; mt < MT; ++mt)
#pragma unroll
            for (int r = 0; r < 8; ++r) {
                int m = mt * 16 + r + hi * 8;
                float v = acc[mt][r] + bias[m] + cond[b * CC + m];
                out[((size_t)b * CC + m) * TT + t] = v;
            }
    } else if constexpr (MODE == 1) {
        // channel LayerNorm: lane L and L^16 jointly hold all 192 rows of col t
        float s = 0.0f, q = 0.0f;
#pragma unroll
        for (int mt = 0; mt < MT; ++mt)
#pragma unroll
            for (int r = 0; r < 8; ++r) {
                int m = mt * 16 + r + hi * 8;
                float v = acc[mt][r] + bias[m];
                acc[mt][r] = v;
                s += v; q += v * v;
            }
        s += __shfl_xor(s, 16, 32);
        q += __shfl_xor(q, 16, 32);
        const float mean = s * (1.0f / 192.0f);
        const float var  = q * (1.0f / 192.0f) - mean * mean;
        const float rstd = rsqrtf(var + 1e-5f);
#pragma unroll
        for (int mt = 0; mt < MT; ++mt)
#pragma unroll
            for (int r = 0; r < 8; ++r) {
                int m = mt * 16 + r + hi * 8;
                float y = lng[m] * (acc[mt][r] - mean) * rstd + lnb[m];
                y = geluf(y);
                size_t oi = ((size_t)b * CC + m) * TT + t;
                out[oi] = resid[oi] + y;
            }
    } else { // MODE 2
        const float mv = outmask ? outmask[b * TT + t] : 1.0f;
#pragma unroll
        for (int mt = 0; mt < MT; ++mt)
#pragma unroll
            for (int r = 0; r < 8; ++r) {
                int m = mt * 16 + r + hi * 8;
                if (m < Mrows) {
                    float v = (acc[mt][r] + bias[m]) * mv;
                    out[((size_t)b * Mrows + m) * TT + t] = v;
                }
            }
    }
}

// ---------------------------------------------------------------------------
// Depthwise conv (k=3, dilation d) + bias + channel-LayerNorm + GELU.
// One block = 128 t-columns of one batch; per-column stats in registers.
// ---------------------------------------------------------------------------
__global__ __launch_bounds__(128) void dw_ln_gelu(
    const float* __restrict__ X, const float* __restrict__ mask,
    const float* __restrict__ dww, const float* __restrict__ dwb,
    const float* __restrict__ g, const float* __restrict__ bta,
    float* __restrict__ Y, int dil)
{
    extern __shared__ float lds[];            // [192][128]
    const int t  = threadIdx.x;
    const int b  = blockIdx.x >> 3;
    const int t0 = (blockIdx.x & 7) << 7;
    const int tg = t0 + t;

    float s = 0.0f, q = 0.0f;
    for (int c = 0; c < CC; ++c) {
        const float* xr = X + ((size_t)b * CC + c) * TT;
        const float w0 = dww[c * 3 + 0], w1 = dww[c * 3 + 1], w2 = dww[c * 3 + 2];
        float v = dwb[c];
        const int tm = tg - dil, tp = tg + dil;
        if (tm >= 0) v += w0 * xr[tm] * mask[b * TT + tm];
        v += w1 * xr[tg] * mask[b * TT + tg];
        if (tp < TT) v += w2 * xr[tp] * mask[b * TT + tp];
        s += v; q += v * v;
        lds[c * 128 + t] = v;
    }
    const float mean = s * (1.0f / 192.0f);
    const float var  = q * (1.0f / 192.0f) - mean * mean;
    const float rstd = rsqrtf(var + 1e-5f);
    for (int c = 0; c < CC; ++c) {
        float y = g[c] * (lds[c * 128 + t] - mean) * rstd + bta[c];
        Y[((size_t)b * CC + c) * TT + tg] = geluf(y);
    }
}

// cond[b,c] = embed[b,:] @ cond_w[c,:] + cond_b[c]
__global__ void k_cond(const float* __restrict__ embed, const float* __restrict__ cw,
                       const float* __restrict__ cb, float* __restrict__ cond)
{
    int idx = blockIdx.x * 256 + threadIdx.x;
    if (idx >= BB * CC) return;
    int b = idx / CC, c = idx - b * CC;
    const float* e  = embed + b * CC;
    const float* wr = cw + c * CC;
    float s = cb[c];
    for (int k = 0; k < CC; ++k) s += e[k] * wr[k];
    cond[idx] = s;
}

__global__ void k_split(const float* __restrict__ noise,
                        float* __restrict__ p0, float* __restrict__ p1)
{
    int idx = blockIdx.x * 256 + threadIdx.x;
    if (idx >= NCOL) return;
    int b = idx / TT, t = idx - b * TT;
    p0[idx] = noise[((size_t)b * 2 + 0) * TT + t];
    p1[idx] = noise[((size_t)b * 2 + 1) * TT + t];
}

// work[b,c,t] = cf_pre_w[c] * x0[b,t] + cf_pre_b[c] + xc[b,c,t]
__global__ void k_flowinit(const float* __restrict__ prew, const float* __restrict__ preb,
                           const float* __restrict__ x0, const float* __restrict__ xc,
                           float* __restrict__ work)
{
    size_t idx = (size_t)blockIdx.x * 256 + threadIdx.x;
    if (idx >= (size_t)BB * CC * TT) return;
    int t = (int)(idx % TT);
    int c = (int)((idx / TT) % CC);
    int b = (int)(idx / ((size_t)CC * TT));
    work[idx] = prew[c] * x0[b * TT + t] + preb[c] + xc[idx];
}

// ---------------------------------------------------------------------------
// Rational-quadratic spline inverse (NBINS=10, TAIL=5). One thread per (b,t).
// Also applies the trailing mask to both channel planes.
// ---------------------------------------------------------------------------
__global__ void k_rqs(const float* __restrict__ hh, const float* __restrict__ mask,
                      float* __restrict__ cur0, float* __restrict__ cur1)
{
    int idx = blockIdx.x * 256 + threadIdx.x;
    if (idx >= NCOL) return;
    const int b = idx / TT, t = idx - b * TT;
    const size_t base = (size_t)b * 29 * TT + t;
    const float inv_s = 0.07216878364870322f;  // 1/sqrt(192)
    const float TAIL = 5.0f, MINW = 0.001f, MINH = 0.001f, MIND = 0.001f;

    float uw[10], uh[10], ud9[9];
#pragma unroll
    for (int m = 0; m < 10; ++m) {
        uw[m] = hh[base + (size_t)m * TT] * inv_s;
        uh[m] = hh[base + (size_t)(10 + m) * TT] * inv_s;
    }
#pragma unroll
    for (int m = 0; m < 9; ++m) ud9[m] = hh[base + (size_t)(20 + m) * TT];

    // widths
    float cw[11], ch[11], dd[11];
    {
        float mx = uw[0];
#pragma unroll
        for (int i = 1; i < 10; ++i) mx = fmaxf(mx, uw[i]);
        float e[10], sum = 0.0f;
#pragma unroll
        for (int i = 0; i < 10; ++i) { e[i] = expf(uw[i] - mx); sum += e[i]; }
        float inv = 1.0f / sum, cs = 0.0f;
        cw[0] = -TAIL;
#pragma unroll
        for (int i = 0; i < 10; ++i) {
            cs += MINW + (1.0f - 10.0f * MINW) * e[i] * inv;
            cw[i + 1] = 2.0f * TAIL * cs - TAIL;
        }
        cw[10] = TAIL;
    }
    // heights
    {
        float mx = uh[0];
#pragma unroll
        for (int i = 1; i < 10; ++i) mx = fmaxf(mx, uh[i]);
        float e[10], sum = 0.0f;
#pragma unroll
        for (int i = 0; i < 10; ++i) { e[i] = expf(uh[i] - mx); sum += e[i]; }
        float inv = 1.0f / sum, cs = 0.0f;
        ch[0] = -TAIL;
#pragma unroll
        for (int i = 0; i < 10; ++i) {
            cs += MINH + (1.0f - 10.0f * MINH) * e[i] * inv;
            ch[i + 1] = 2.0f * TAIL * cs - TAIL;
        }
        ch[10] = TAIL;
    }
    // derivatives (padded ends -> softplus(log(expm1(1-MIND))) = 1-MIND, so d=1.0)
    dd[0] = 1.0f; dd[10] = 1.0f;
#pragma unroll
    for (int i = 1; i < 10; ++i) dd[i] = MIND + softplusf(ud9[i - 1]);

    const float inp = cur1[idx];
    const bool inside = (inp >= -TAIL) && (inp <= TAIL);
    const float xcv = fminf(fmaxf(inp, -TAIL), TAIL);

    int cnt = 0;
#pragma unroll
    for (int j = 0; j < 11; ++j) {
        float loc = ch[j] + ((j == 10) ? 1e-6f : 0.0f);
        cnt += (xcv >= loc) ? 1 : 0;
    }
    int bi = min(max(cnt - 1, 0), 9);

    const float bw = cw[bi + 1] - cw[bi], bcw = cw[bi];
    const float bh = ch[bi + 1] - ch[bi], bch = ch[bi];
    const float d0 = dd[bi], d1 = dd[bi + 1];
    const float delta = bh / bw;
    const float two = d0 + d1 - 2.0f * delta;
    const float y = xcv - bch;
    const float a  = y * two + bh * (delta - d0);
    const float bb = bh * d0 - y * two;
    const float c2 = -delta * y;
    const float disc = fmaxf(bb * bb - 4.0f * a * c2, 0.0f);
    const float root = 2.0f * c2 / (-bb - sqrtf(disc));
    const float outv = root * bw + bcw;

    const float mv = mask[idx];
    cur1[idx] = (inside ? outv : inp) * mv;
    cur0[idx] = cur0[idx] * mv;
}

__global__ void k_final(const float* __restrict__ z0, const float* __restrict__ ea_m,
                        const float* __restrict__ ea_logs, const float* __restrict__ mask,
                        float* __restrict__ out)
{
    int idx = blockIdx.x * 256 + threadIdx.x;
    if (idx >= NCOL) return;
    out[idx] = (z0[idx] - ea_m[0]) * expf(-ea_logs[0]) * mask[idx];
}

// ---------------------------------------------------------------------------
extern "C" void kernel_launch(void* const* d_in, const int* in_sizes, int n_in,
                              void* d_out, int out_size, void* d_ws, size_t ws_size,
                              hipStream_t stream)
{
    (void)in_sizes; (void)n_in; (void)out_size; (void)ws_size;
    const float* x        = (const float*)d_in[0];
    const float* mask     = (const float*)d_in[1];
    const float* embed    = (const float*)d_in[2];
    const float* noise    = (const float*)d_in[3];
    const float* pre_w    = (const float*)d_in[4];
    const float* pre_b    = (const float*)d_in[5];
    const float* cond_w   = (const float*)d_in[6];
    const float* cond_b   = (const float*)d_in[7];
    const float* dds_dw_w = (const float*)d_in[8];
    const float* dds_dw_b = (const float*)d_in[9];
    const float* dds_pw_w = (const float*)d_in[10];
    const float* dds_pw_b = (const float*)d_in[11];
    const float* dds_n1_g = (const float*)d_in[12];
    const float* dds_n1_b = (const float*)d_in[13];
    const float* dds_n2_g = (const float*)d_in[14];
    const float* dds_n2_b = (const float*)d_in[15];
    const float* proj_w   = (const float*)d_in[16];
    const float* proj_b   = (const float*)d_in[17];
    const float* ea_m     = (const float*)d_in[18];
    const float* ea_logs  = (const float*)d_in[19];
    const float* cf_pre_w = (const float*)d_in[20];
    const float* cf_pre_b = (const float*)d_in[21];
    const float* cf_dw_w  = (const float*)d_in[22];
    const float* cf_dw_b  = (const float*)d_in[23];
    const float* cf_pw_w  = (const float*)d_in[24];
    const float* cf_pw_b  = (const float*)d_in[25];
    const float* cf_n1_g  = (const float*)d_in[26];
    const float* cf_n1_b  = (const float*)d_in[27];
    const float* cf_n2_g  = (const float*)d_in[28];
    const float* cf_n2_b  = (const float*)d_in[29];
    const float* cf_proj_w= (const float*)d_in[30];
    const float* cf_proj_b= (const float*)d_in[31];
    float* out = (float*)d_out;

    const size_t S = (size_t)BB * CC * TT;  // 12,582,912 floats
    float* ws     = (float*)d_ws;
    float* xcbuf  = ws;               // conditioning g
    float* work   = ws + S;           // running dds state
    float* tmpA   = ws + 2 * S;       // dw+LN+GELU output (GEMM input)
    float* hhb    = ws + 3 * S;       // (B,29,T) spline params
    float* plane0 = hhb + (size_t)BB * 29 * TT;
    float* plane1 = plane0 + NCOL;
    float* condb  = plane1 + NCOL;    // (B,C)

    const size_t SH12 = (size_t)(12 * 16 + 128) * KPAD * 2;  // 133120 B
    const size_t SH2  = (size_t)(2 * 16 + 128) * KPAD * 2;   // 66560 B
    const size_t SHDW = (size_t)CC * 128 * 4;                // 98304 B
    const dim3 gG(NCOL / 128), bG(256);
    const dim3 gDW(BB * (TT / 128)), bDW(128);
    const dim3 gPT((NCOL + 255) / 256), bPT(256);

    // --- conditioning path: xc ---
    k_cond<<<dim3((BB * CC + 255) / 256), bPT, 0, stream>>>(embed, cond_w, cond_b, condb);
    gemm192<12, 0><<<gG, bG, SH12, stream>>>(pre_w, CC, x, pre_b, condb,
        nullptr, nullptr, nullptr, nullptr, nullptr, work);
    int dil = 1;
    for (int l = 0; l < 3; ++l, dil *= 3) {
        dw_ln_gelu<<<gDW, bDW, SHDW, stream>>>(work, mask,
            dds_dw_w + l * CC * 3, dds_dw_b + l * CC,
            dds_n1_g + l * CC, dds_n1_b + l * CC, tmpA, dil);
        gemm192<12, 1><<<gG, bG, SH12, stream>>>(dds_pw_w + (size_t)l * CC * CC, CC,
            tmpA, dds_pw_b + l * CC, nullptr, work,
            dds_n2_g + l * CC, dds_n2_b + l * CC, nullptr, nullptr, work);
    }
    gemm192<12, 2><<<gG, bG, SH12, stream>>>(proj_w, CC, work, proj_b,
        nullptr, nullptr, nullptr, nullptr, mask, mask, xcbuf);

    // --- flows (i = 3,2,1), channel flip == plane pointer swap ---
    k_split<<<gPT, bPT, 0, stream>>>(noise, plane0, plane1);
    float* cur0 = plane0;
    float* cur1 = plane1;
    for (int i = 3; i >= 1; --i) {
        float* tmp = cur0; cur0 = cur1; cur1 = tmp;  // z = z[:, ::-1]
        k_flowinit<<<dim3((unsigned)((S + 255) / 256)), bPT, 0, stream>>>(
            cf_pre_w + i * CC, cf_pre_b + i * CC, cur0, xcbuf, work);
        dil = 1;
        for (int l = 0; l < 3; ++l, dil *= 3) {
            const size_t il = (size_t)i * 3 + l;
            dw_ln_gelu<<<gDW, bDW, SHDW, stream>>>(work, mask,
                cf_dw_w + il * CC * 3, cf_dw_b + il * CC,
                cf_n1_g + il * CC, cf_n1_b + il * CC, tmpA, dil);
            gemm192<12, 1><<<gG, bG, SH12, stream>>>(cf_pw_w + il * CC * CC, CC,
                tmpA, cf_pw_b + il * CC, nullptr, work,
                cf_n2_g + il * CC, cf_n2_b + il * CC, nullptr, nullptr, work);
        }
        gemm192<2, 2><<<gG, bG, SH2, stream>>>(cf_proj_w + (size_t)i * 29 * CC, 29,
            work, cf_proj_b + i * 29, nullptr, nullptr, nullptr, nullptr,
            mask, mask, hhb);
        k_rqs<<<gPT, bPT, 0, stream>>>(hhb, mask, cur0, cur1);
    }
    { float* tmp = cur0; cur0 = cur1; cur1 = tmp; }  // final flip
    k_final<<<gPT, bPT, 0, stream>>>(cur0, ea_m, ea_logs, mask, out);
}
